// ResGCN_25666724560908
// MI455X (gfx1250) — compile-verified, matrix operand order checked by reading, not run on hardware
//
#include <hip/hip_runtime.h>
#include <hip/hip_bf16.h>
#include <math.h>

typedef __bf16 bf16_t;
typedef __attribute__((ext_vector_type(16))) __bf16 v16bf;
typedef __attribute__((ext_vector_type(8)))  float  v8f;

#define D_FEAT 128
#define TPB 256

// ---------------- utility: fill ----------------
__global__ void k_fill(float* __restrict__ p, float v, int count) {
  int i = blockIdx.x * blockDim.x + threadIdx.x;
  if (i < count) p[i] = v;
}

// ---------------- degree / norm precompute ----------------
__global__ void k_deg_scatter(const int* __restrict__ dst, const float* __restrict__ w,
                              float* __restrict__ deg, int E) {
  int e = blockIdx.x * blockDim.x + threadIdx.x;
  if (e < E) atomicAdd(&deg[dst[e]], w[e]);
}

__global__ void k_dinv(float* __restrict__ deg, int n) {
  int i = blockIdx.x * blockDim.x + threadIdx.x;
  if (i < n) {
    float d = deg[i];
    deg[i] = (d > 0.0f) ? rsqrtf(d) : 0.0f;
  }
}

__global__ void k_norm(const int* __restrict__ src, const int* __restrict__ dst,
                       const float* __restrict__ w, const float* __restrict__ dinv,
                       float* __restrict__ normv, int E, int N) {
  int e = blockIdx.x * blockDim.x + threadIdx.x;
  if (e < E) {
    normv[e] = dinv[src[e]] * w[e] * dinv[dst[e]];
  } else if (e < E + N) {
    int i = e - E;
    normv[e] = dinv[i] * dinv[i];   // self loop, weight 1
  }
}

// ---------------- weight packing into WMMA B-fragment order ----------------
// Fragment (kt, nt): 32 lanes x 16 bf16, contiguous. Lane l: column n = nt*16 + (l&15),
// half = l>>4; element j holds K = kt*32 + half*8 + (j&7) + (j>=8 ? 16 : 0).
__global__ void k_pack_w(const float* __restrict__ W, bf16_t* __restrict__ Wp,
                         int ncols_src, int ntiles) {
  int t = blockIdx.x * blockDim.x + threadIdx.x;
  int total = 4 * ntiles * 32;            // 4 K-tiles (K = 128)
  if (t >= total) return;
  int lane = t & 31;
  int frag = t >> 5;
  int nt   = frag % ntiles;
  int kt   = frag / ntiles;
  int half = lane >> 4;
  int n    = nt * 16 + (lane & 15);
  bf16_t* dstp = Wp + (size_t)t * 16;
#pragma unroll
  for (int j = 0; j < 16; ++j) {
    int K = kt * 32 + half * 8 + (j & 7) + ((j >= 8) ? 16 : 0);
    float v = (n < ncols_src) ? W[(size_t)K * ncols_src + n] : 0.0f;
    dstp[j] = (bf16_t)v;
  }
}

// ---------------- GEMM: H[n x (NTILES*16)] = X[n x 128] @ W(packed) ----------------
// 8 waves/block, one 16-row M-tile per wave. Packed weights staged once per block
// into LDS with CDNA5 async global->LDS copies; fragments read via ds_load_b128 with
// immediate offsets (NTILES/LDH are compile-time), fully unrolled: 32 WMMAs straight-line.
template <int NTILES, int LDH>
__global__ __launch_bounds__(TPB)
void k_gemm_bf16(const float* __restrict__ X, const bf16_t* __restrict__ Wp,
                 float* __restrict__ H, int n) {
  __shared__ __align__(64) bf16_t wlds[4 * NTILES * 32 * 16];

  const int tid  = threadIdx.x;
  const int lane = tid & 31;
  constexpr int NBYTES = 4 * NTILES * 32 * 16 * (int)sizeof(bf16_t);  // 16KB or 32KB

  // ---- async stage packed weights into LDS (all threads participate) ----
  {
    unsigned lds_base = (unsigned)(uintptr_t)(void*)wlds;   // aperture low bits = LDS offset
    const char* gsrc = (const char*)Wp;
#pragma unroll
    for (int ofs = 0; ofs < NBYTES; ofs += TPB * 16) {
      unsigned      ldst  = lds_base + (unsigned)(ofs + tid * 16);
      unsigned long gaddr = (unsigned long)(gsrc + ofs + tid * 16);
      asm volatile("global_load_async_to_lds_b128 %0, %1, off"
                   :: "v"(ldst), "v"(gaddr) : "memory");
    }
    asm volatile("s_wait_asynccnt 0" ::: "memory");
    __syncthreads();   // every wave waited for its own async writes first
  }

  const int mtile = (blockIdx.x * blockDim.x + tid) >> 5;
  if (mtile * 16 >= n) return;

  const int half = lane >> 4;
  const int row  = mtile * 16 + (lane & 15);
  const int rowc = (row < n) ? row : (n - 1);
  const float* xr = X + (size_t)rowc * D_FEAT;

  // Build 4 A fragments (16x32 bf16 each) per ISA 16-bit A layout.
  v16bf a[4];
#pragma unroll
  for (int kt = 0; kt < 4; ++kt) {
    int kb = kt * 32 + half * 8;
#pragma unroll
    for (int j = 0; j < 8; ++j) {
      a[kt][j]     = (bf16_t)xr[kb + j];
      a[kt][8 + j] = (bf16_t)xr[kb + 16 + j];
    }
  }

  const bool full    = (mtile * 16 + 16 <= n);
  const int  rowbase = mtile * 16 + half * 8;
  const bf16_t* lbase = wlds + (size_t)lane * 16;   // per-lane base; rest is immediate offsets

#pragma unroll
  for (int nt = 0; nt < NTILES; ++nt) {
    v8f acc = {};
#pragma unroll
    for (int kt = 0; kt < 4; ++kt) {
      const v16bf* bp =
          (const v16bf*)(lbase + (size_t)(kt * NTILES + nt) * 32 * 16);
      v16bf b = *bp;                                    // 2x ds_load_b128, imm offsets
      acc = __builtin_amdgcn_wmma_f32_16x16x32_bf16(
          /*neg_a=*/false, a[kt], /*neg_b=*/false, b,
          /*c_mod=*/(short)0, acc, /*reuse_a=*/false, /*reuse_b=*/false);
    }
    // C/D layout: value (M = half*8 + r, Ncol = lane&15) in VGPR r.
    float* hp = H + (size_t)rowbase * LDH + (nt * 16 + (lane & 15));
    if (full) {
#pragma unroll
      for (int r = 0; r < 8; ++r)
        hp[(size_t)r * LDH] = acc[r];                   // imm offsets r*LDH*4
    } else {
#pragma unroll
      for (int r = 0; r < 8; ++r)
        if (rowbase + r < n) hp[(size_t)r * LDH] = acc[r];
    }
  }
}

// ---------------- edge aggregation: O[dst] += H[src] * norm ----------------
// One wave per edge (incl. self loops appended logically at the end).
template <int F, int LD>
__global__ __launch_bounds__(TPB)
void k_aggregate(const float* __restrict__ H, float* __restrict__ O,
                 const int* __restrict__ src, const int* __restrict__ dst,
                 const float* __restrict__ normv, int E, int total_edges) {
  int wave = (blockIdx.x * blockDim.x + threadIdx.x) >> 5;
  int lane = threadIdx.x & 31;
  if (wave >= total_edges) return;
  int s, d;
  if (wave < E) { s = src[wave]; d = dst[wave]; }
  else          { s = d = wave - E; }
  float nv = normv[wave];
  const float* hs = H + (size_t)s * LD;
  float*       od = O + (size_t)d * LD;
  if constexpr (F == D_FEAT) {
    // one float4 per lane: b128 load + 4 f32 atomics
    float4 hv = ((const float4*)hs)[lane];
    int f = lane * 4;
    atomicAdd(&od[f + 0], hv.x * nv);
    atomicAdd(&od[f + 1], hv.y * nv);
    atomicAdd(&od[f + 2], hv.z * nv);
    atomicAdd(&od[f + 3], hv.w * nv);
  } else {
#pragma unroll
    for (int f0 = 0; f0 < F; f0 += 32) {
      int f = f0 + lane;
      if (f < F) atomicAdd(&od[f], hs[f] * nv);
    }
  }
}

// ---------------- bias + relu (in place), F == 128 ----------------
__global__ void k_bias_relu(float* __restrict__ A, const float* __restrict__ b, int count) {
  int i = blockIdx.x * blockDim.x + threadIdx.x;
  if (i >= count) return;
  int f = i & (D_FEAT - 1);
  float v = A[i] + b[f];
  A[i] = (v > 0.0f) ? v : 0.0f;
}

// ---------------- block epilogue: out = (L2norm(relu(agg+b)) + identity)/2 ----------------
// One wave per row; OUT may alias AGG.
__global__ __launch_bounds__(TPB)
void k_post2(const float* __restrict__ AGG, const float* __restrict__ bias,
             const float* __restrict__ IDENT, float* __restrict__ OUT, int n) {
  int wave = (blockIdx.x * blockDim.x + threadIdx.x) >> 5;
  int lane = threadIdx.x & 31;
  if (wave >= n) return;
  const float* ar = AGG   + (size_t)wave * D_FEAT;
  const float* ir = IDENT + (size_t)wave * D_FEAT;
  float*       orow = OUT + (size_t)wave * D_FEAT;
  float v[4];
  float ss = 0.0f;
#pragma unroll
  for (int j = 0; j < 4; ++j) {
    float t = ar[lane + j * 32] + bias[lane + j * 32];
    t = (t > 0.0f) ? t : 0.0f;
    v[j] = t;
    ss += t * t;
  }
  for (int off = 16; off; off >>= 1) ss += __shfl_xor(ss, off, 32);
  float inv = 1.0f / fmaxf(sqrtf(ss), 1e-12f);
#pragma unroll
  for (int j = 0; j < 4; ++j)
    orow[lane + j * 32] = (v[j] * inv + ir[lane + j * 32]) * 0.5f;
}

// ---------------- final: log_softmax(agg + bias) over 40 classes ----------------
__global__ __launch_bounds__(TPB)
void k_logsoftmax(const float* __restrict__ AGG, const float* __restrict__ bias,
                  float* __restrict__ OUT, int n) {
  const int C = 40, LD = 64;
  int wave = (blockIdx.x * blockDim.x + threadIdx.x) >> 5;
  int lane = threadIdx.x & 31;
  if (wave >= n) return;
  const float* ar = AGG + (size_t)wave * LD;
  float z0 = (lane < C)      ? (ar[lane]      + bias[lane])      : -1e30f;
  float z1 = (lane + 32 < C) ? (ar[lane + 32] + bias[lane + 32]) : -1e30f;
  float m = fmaxf(z0, z1);
  for (int off = 16; off; off >>= 1) m = fmaxf(m, __shfl_xor(m, off, 32));
  float e = ((lane < C) ? expf(z0 - m) : 0.0f) +
            ((lane + 32 < C) ? expf(z1 - m) : 0.0f);
  for (int off = 16; off; off >>= 1) e += __shfl_xor(e, off, 32);
  float lse = m + logf(e);
  if (lane < C)      OUT[(size_t)wave * C + lane]      = z0 - lse;
  if (lane + 32 < C) OUT[(size_t)wave * C + lane + 32] = z1 - lse;
}

// ---------------- host orchestration ----------------
static inline int ceil_div(long a, int b) { return (int)((a + b - 1) / b); }

extern "C" void kernel_launch(void* const* d_in, const int* in_sizes, int n_in,
                              void* d_out, int out_size, void* d_ws, size_t ws_size,
                              hipStream_t stream) {
  const float* x  = (const float*)d_in[0];
  const int*   ei = (const int*)  d_in[1];
  const float* ew = (const float*)d_in[2];
  const float* Wb = (const float*)d_in[3];
  const float* bb = (const float*)d_in[4];
  const float* Wf = (const float*)d_in[5];
  const float* bf = (const float*)d_in[6];

  const int N = in_sizes[0] / D_FEAT;
  const int E = in_sizes[2];
  const int TE = E + N;                  // edges + self loops
  const int* srcI = ei;
  const int* dstI = ei + E;

  // workspace carve-out (256B aligned)
  char* wsb = (char*)d_ws;
  size_t off = 0;
  auto carve = [&](size_t bytes) -> void* {
    off = (off + 255) & ~(size_t)255;
    void* p = wsb + off;
    off += bytes;
    return p;
  };
  float*  normv = (float*) carve((size_t)TE * 4);
  float*  dinv  = (float*) carve((size_t)N * 4);
  // packed weights: 8 block mats (4kt*8nt*32*16 bf16 = 16384) + final (4kt*4nt*32*16 = 8192)
  bf16_t* Wpk   = (bf16_t*)carve((size_t)(8 * 16384 + 8192) * sizeof(bf16_t));
  float*  B0    = (float*) carve((size_t)N * D_FEAT * 4);
  float*  B1    = (float*) carve((size_t)N * D_FEAT * 4);
  float*  B2    = (float*) carve((size_t)N * D_FEAT * 4);

  // ---- one-time normalization coefficients ----
  k_fill<<<ceil_div(N, TPB), TPB, 0, stream>>>(dinv, 1.0f, N);      // self loop wt
  k_deg_scatter<<<ceil_div(E, TPB), TPB, 0, stream>>>(dstI, ew, dinv, E);
  k_dinv<<<ceil_div(N, TPB), TPB, 0, stream>>>(dinv, N);
  k_norm<<<ceil_div(TE, TPB), TPB, 0, stream>>>(srcI, dstI, ew, dinv, normv, E, N);

  // ---- pack all weights to bf16 fragments ----
  for (int m = 0; m < 8; ++m)
    k_pack_w<<<ceil_div(4 * 8 * 32, TPB), TPB, 0, stream>>>(
        Wb + (size_t)m * D_FEAT * D_FEAT, Wpk + (size_t)m * 16384, D_FEAT, 8);
  k_pack_w<<<ceil_div(4 * 4 * 32, TPB), TPB, 0, stream>>>(
      Wf, Wpk + (size_t)8 * 16384, 40, 4);

  const int gemm_blocks = ceil_div((long)ceil_div(N, 16) * 32, TPB);
  const int aggr_blocks = ceil_div((long)TE * 32, TPB);
  const int elem_blocks = ceil_div((long)N * D_FEAT, TPB);
  const int row_blocks  = ceil_div((long)N * 32, TPB);

  // ---- 4 residual blocks ----
  auto run_block = [&](const float* xin, float* s1, float* s2,
                       const bf16_t* W1, const float* b1,
                       const bf16_t* W2, const float* b2) {
    k_gemm_bf16<8, D_FEAT><<<gemm_blocks, TPB, 0, stream>>>(xin, W1, s1, N);
    k_fill<<<elem_blocks, TPB, 0, stream>>>(s2, 0.0f, N * D_FEAT);
    k_aggregate<D_FEAT, D_FEAT><<<aggr_blocks, TPB, 0, stream>>>(s1, s2, srcI, dstI,
                                                                 normv, E, TE);
    k_bias_relu<<<elem_blocks, TPB, 0, stream>>>(s2, b1, N * D_FEAT);
    k_gemm_bf16<8, D_FEAT><<<gemm_blocks, TPB, 0, stream>>>(s2, W2, s1, N);
    k_fill<<<elem_blocks, TPB, 0, stream>>>(s2, 0.0f, N * D_FEAT);
    k_aggregate<D_FEAT, D_FEAT><<<aggr_blocks, TPB, 0, stream>>>(s1, s2, srcI, dstI,
                                                                 normv, E, TE);
    k_post2<<<row_blocks, TPB, 0, stream>>>(s2, b2, xin, s2, N);   // result in s2
  };

  run_block(x,  B0, B1, Wpk + 0 * 16384, bb + 0 * D_FEAT, Wpk + 1 * 16384, bb + 1 * D_FEAT);
  run_block(B1, B0, B2, Wpk + 2 * 16384, bb + 2 * D_FEAT, Wpk + 3 * 16384, bb + 3 * D_FEAT);
  run_block(B2, B0, B1, Wpk + 4 * 16384, bb + 4 * D_FEAT, Wpk + 5 * 16384, bb + 5 * D_FEAT);
  run_block(B1, B0, B2, Wpk + 6 * 16384, bb + 6 * D_FEAT, Wpk + 7 * 16384, bb + 7 * D_FEAT);

  // ---- final conv (D=128 -> 40, padded to 64 cols) + log_softmax ----
  k_gemm_bf16<4, 64><<<gemm_blocks, TPB, 0, stream>>>(B2, Wpk + (size_t)8 * 16384, B0, N);
  k_fill<<<ceil_div((long)N * 64, TPB), TPB, 0, stream>>>(B1, 0.0f, N * 64);
  k_aggregate<40, 64><<<aggr_blocks, TPB, 0, stream>>>(B0, B1, srcI, dstI, normv, E, TE);
  k_logsoftmax<<<row_blocks, TPB, 0, stream>>>(B1, bf, (float*)d_out, N);
}
